// TalkingHeadsAttention_33792802685367
// MI455X (gfx1250) — compile-verified, matrix operand order checked by reading, not run on hardware
//
#include <hip/hip_runtime.h>

// ---------------------------------------------------------------------------
// Talking-heads attention for gfx1250 (MI455X), bf16 WMMA pipeline.
// B=4, S=1024, D=768, H=12, HD=64.
// ---------------------------------------------------------------------------

typedef __attribute__((ext_vector_type(16))) __bf16 v16bf;
typedef __attribute__((ext_vector_type(8)))  __bf16 v8bf;
typedef __attribute__((ext_vector_type(8)))  float  v8f;

#define D_MODEL 768
#define NHEAD   12
#define HDIM    64
#define SEQ     1024
#define BATCH   4
#define MROWS   (BATCH * SEQ)   // 4096

__device__ __forceinline__ __bf16 f2bf(float f) {
  union { float f; unsigned u; } a; a.f = f;
  unsigned u = a.u;
  u += 0x7fffu + ((u >> 16) & 1u);               // round to nearest even
  union { unsigned short s; __bf16 b; } r;
  r.s = (unsigned short)(u >> 16);
  return r.b;
}

// Load one 16x32 (rows x K) bf16 operand tile in the CDNA5 WMMA A/B lane
// layout from row-major memory (stride in elements).
//   lane l: row = l&15 ; K chunks = {sel*8 .. sel*8+7} and {16+sel*8 ..}
__device__ __forceinline__ v16bf load_ab_tile(const __bf16* base, int stride, int lane) {
  const int r   = lane & 15;
  const int sel = lane >> 4;
  const __bf16* p = base + (size_t)r * stride + sel * 8;
  v8bf lo = *(const v8bf*)(p);
  v8bf hi = *(const v8bf*)(p + 16);
  v16bf v;
#pragma unroll
  for (int i = 0; i < 8; ++i) { v[i] = lo[i]; v[8 + i] = hi[i]; }
  return v;
}

__device__ __forceinline__ v8f wmma_bf16(v16bf a, v16bf b, v8f c) {
  return __builtin_amdgcn_wmma_f32_16x16x32_bf16(false, a, false, b, (short)0, c,
                                                 false, false);
}

// ---------------------------------------------------------------------------
// Kernel 0: fp32 -> bf16 conversion
// ---------------------------------------------------------------------------
__global__ void k_cvt_bf16(const float* __restrict__ src, __bf16* __restrict__ dst, int n) {
  int i = blockIdx.x * blockDim.x + threadIdx.x;
  if (i < n) dst[i] = f2bf(src[i]);
}

// ---------------------------------------------------------------------------
// Kernel 1: projection GEMM  Y = X @ Wq^T + bq, split into heads.
//   mode 0/1 : store row-major into dst[b][h][s][d]   (qh / kh)
//   mode 2   : store transposed into dst[b][h][d][s]  (vhT)
// One wave per 16x16 output tile.  K loop: 768 / 32 = 24 WMMAs.
// ---------------------------------------------------------------------------
__global__ void k_proj(const __bf16* __restrict__ X, const __bf16* __restrict__ W,
                       const float* __restrict__ bias, __bf16* __restrict__ dst,
                       int mode) {
  const int mt = blockIdx.x;          // 0..255
  const int nt = blockIdx.y;          // 0..47
  const int lane = threadIdx.x;       // 0..31
  const int m0 = mt * 16, n0 = nt * 16;

  v8f acc = {0.f, 0.f, 0.f, 0.f, 0.f, 0.f, 0.f, 0.f};
  for (int k = 0; k < D_MODEL; k += 32) {
    v16bf a = load_ab_tile(X + (size_t)m0 * D_MODEL + k, D_MODEL, lane);
    v16bf b = load_ab_tile(W + (size_t)n0 * D_MODEL + k, D_MODEL, lane);
    acc = wmma_bf16(a, b, acc);
  }

  const int n = n0 + (lane & 15);
  const float bv = bias[n];
  const int mbase = m0 + 8 * (lane >> 4);
  const int h = n >> 6;               // head
  const int d = n & 63;               // dim in head

  if (mode < 2) {
    // dst[b][h][s][d]
#pragma unroll
    for (int j = 0; j < 8; ++j) {
      int m = mbase + j;
      int b_ = m >> 10, s = m & 1023;
      dst[(((size_t)(b_ * NHEAD + h)) * SEQ + s) * HDIM + d] = f2bf(acc[j] + bv);
    }
  } else {
    // vhT[b][h][d][s] : 8 contiguous s per lane -> one 16B store
    int b_ = mbase >> 10, s = mbase & 1023;
    __bf16* p = dst + (((size_t)(b_ * NHEAD + h)) * HDIM + d) * SEQ + s;
    v8bf o;
#pragma unroll
    for (int j = 0; j < 8; ++j) o[j] = f2bf(acc[j] + bv);
    *(v8bf*)p = o;
  }
}

// ---------------------------------------------------------------------------
// Kernel 2: logits + pre-softmax talking-heads mix.
//   Lmix[b][g][q][k] = sum_h (Wl[g,h]/8) * (qh[b,h,q,:] . kh[b,h,k,:]) + bl[g]
// One wave per (b, qtile, ktile); 12 heads x 2 WMMAs inside.
// ---------------------------------------------------------------------------
__global__ void k_logits(const __bf16* __restrict__ qh, const __bf16* __restrict__ kh,
                         const float* __restrict__ Wl, const float* __restrict__ bl,
                         float* __restrict__ Lmix) {
  const int kt = blockIdx.x, qt = blockIdx.y, b = blockIdx.z;
  const int lane = threadIdx.x;

  v8f acc[NHEAD];
#pragma unroll
  for (int g = 0; g < NHEAD; ++g)
#pragma unroll
    for (int j = 0; j < 8; ++j) acc[g][j] = 0.f;

  for (int h = 0; h < NHEAD; ++h) {
    const __bf16* qb = qh + (((size_t)(b * NHEAD + h)) * SEQ + qt * 16) * HDIM;
    const __bf16* kb = kh + (((size_t)(b * NHEAD + h)) * SEQ + kt * 16) * HDIM;
    v16bf a0 = load_ab_tile(qb,      HDIM, lane);
    v16bf a1 = load_ab_tile(qb + 32, HDIM, lane);
    v16bf b0 = load_ab_tile(kb,      HDIM, lane);
    v16bf b1 = load_ab_tile(kb + 32, HDIM, lane);
    v8f raw = {0.f, 0.f, 0.f, 0.f, 0.f, 0.f, 0.f, 0.f};
    raw = wmma_bf16(a0, b0, raw);
    raw = wmma_bf16(a1, b1, raw);
#pragma unroll
    for (int g = 0; g < NHEAD; ++g) {
      float w = Wl[g * NHEAD + h] * 0.125f;   // fold 1/sqrt(64)
#pragma unroll
      for (int j = 0; j < 8; ++j) acc[g][j] += w * raw[j];
    }
  }

  const int n = kt * 16 + (lane & 15);
  const int mbase = qt * 16 + 8 * (lane >> 4);
#pragma unroll
  for (int g = 0; g < NHEAD; ++g) {
    float blg = bl[g];
    float* outp = Lmix + (((size_t)(b * NHEAD + g)) * SEQ + mbase) * SEQ + n;
#pragma unroll
    for (int j = 0; j < 8; ++j) outp[(size_t)j * SEQ] = acc[g][j] + blg;
  }
}

// ---------------------------------------------------------------------------
// Kernel 3: softmax row stats (max, sum-exp) per (b, g, q).
// 256 threads: 16 rows x 16 threads, each thread scans 64 strided elements.
// ---------------------------------------------------------------------------
__global__ void k_stats(const float* __restrict__ Lmix,
                        float* __restrict__ M, float* __restrict__ Z) {
  const int qt = blockIdx.x, g = blockIdx.y, b = blockIdx.z;
  const int tid = threadIdx.x;
  const int r = tid >> 4, sub = tid & 15;
  const int q = qt * 16 + r;
  const float* row = Lmix + (((size_t)(b * NHEAD + g)) * SEQ + q) * SEQ;

  __shared__ float red[16][17];

  float mx = -1e30f;
  for (int j = 0; j < SEQ / 16; ++j) mx = fmaxf(mx, row[sub + j * 16]);
  red[r][sub] = mx;
  __syncthreads();
  if (sub == 0) {
    float m2 = red[r][0];
    for (int t = 1; t < 16; ++t) m2 = fmaxf(m2, red[r][t]);
    red[r][16] = m2;
  }
  __syncthreads();
  const float rowmax = red[r][16];

  float s = 0.f;
  for (int j = 0; j < SEQ / 16; ++j) s += __expf(row[sub + j * 16] - rowmax);
  __syncthreads();
  red[r][sub] = s;
  __syncthreads();
  if (sub == 0) {
    float z2 = 0.f;
    for (int t = 0; t < 16; ++t) z2 += red[r][t];
    size_t idx = ((size_t)(b * NHEAD + g)) * SEQ + q;
    M[idx] = rowmax;
    Z[idx] = z2;
  }
}

// ---------------------------------------------------------------------------
// Kernel 4: fused probs -> post-softmax mix (Wp, bp) -> attn @ V.
// Workgroup = 12 waves (one head each), per (b, qtile).
// Each k-step (32 keys): waves exp() their head's 16x32 prob tile into LDS,
// barrier, mix across heads in fp32, then 4 bf16 WMMAs against vhT.
// bp bias is folded into the mixed probs (each (q,k) appears exactly once).
// ---------------------------------------------------------------------------
__global__ void __launch_bounds__(384)
k_attn(const float* __restrict__ Lmix, const float* __restrict__ M,
       const float* __restrict__ Z, const __bf16* __restrict__ vhT,
       const float* __restrict__ Wp, const float* __restrict__ bp,
       __bf16* __restrict__ outh) {
  const int qt = blockIdx.x, b = blockIdx.y;
  const int w = threadIdx.x >> 5;     // head 0..11 (== output head)
  const int lane = threadIdx.x & 31;
  const int q0 = qt * 16;
  const int r = lane & 15, sel = lane >> 4;

  __shared__ float pbuf[NHEAD][16][32];

  const size_t sidx = ((size_t)(b * NHEAD + w)) * SEQ + q0 + r;
  const float mrow = M[sidx];
  const float invZ = 1.0f / Z[sidx];

  float wp[NHEAD];
#pragma unroll
  for (int g = 0; g < NHEAD; ++g) wp[g] = Wp[w * NHEAD + g];
  const float bpw = bp[w];

  v8f acc[4];
#pragma unroll
  for (int t = 0; t < 4; ++t)
#pragma unroll
    for (int j = 0; j < 8; ++j) acc[t][j] = 0.f;

  const float* Lbase = Lmix + (((size_t)(b * NHEAD + w)) * SEQ + q0 + r) * SEQ;
  const __bf16* Vbase = vhT + ((size_t)(b * NHEAD + w)) * HDIM * SEQ;

  for (int k0 = 0; k0 < SEQ; k0 += 32) {
    // ---- probs for this wave's head: 16 logits for row r, k = k0+sel*16+.. --
    const float4* lp4 = (const float4*)(Lbase + k0 + sel * 16);
    float lv[16];
#pragma unroll
    for (int t = 0; t < 4; ++t) {
      float4 f = lp4[t];
      lv[4 * t + 0] = f.x; lv[4 * t + 1] = f.y;
      lv[4 * t + 2] = f.z; lv[4 * t + 3] = f.w;
    }
    float* pd = &pbuf[w][r][sel * 16];
#pragma unroll
    for (int j = 0; j < 16; ++j) pd[j] = __expf(lv[j] - mrow) * invZ;
    __syncthreads();

    // ---- post-softmax mix, built directly in WMMA-A lane layout ----
    float pp[16];
#pragma unroll
    for (int j = 0; j < 16; ++j) pp[j] = bpw;
    for (int g = 0; g < NHEAD; ++g) {
      const float wg = wp[g];
      const float* pg = &pbuf[g][r][0];
#pragma unroll
      for (int j = 0; j < 8; ++j) {
        pp[j]     += wg * pg[sel * 8 + j];
        pp[8 + j] += wg * pg[16 + sel * 8 + j];
      }
    }
    v16bf a;
#pragma unroll
    for (int j = 0; j < 16; ++j) a[j] = f2bf(pp[j]);

    // ---- attn @ V : four 16x16 output tiles over d ----
#pragma unroll
    for (int t = 0; t < 4; ++t) {
      v16bf bm = load_ab_tile(Vbase + (size_t)(t * 16) * SEQ + k0, SEQ, lane);
      acc[t] = wmma_bf16(a, bm, acc[t]);
    }
    __syncthreads();
  }

  // store: outh[b*S + s][w*64 + t*16 + n] bf16
  const int mb = 8 * sel;
#pragma unroll
  for (int t = 0; t < 4; ++t) {
    const int ch = w * HDIM + t * 16 + r;
#pragma unroll
    for (int j = 0; j < 8; ++j) {
      outh[((size_t)(b * SEQ + q0 + mb + j)) * D_MODEL + ch] = f2bf(acc[t][j]);
    }
  }
}

// ---------------------------------------------------------------------------
// Kernel 5: final projection  out = outh @ Wf^T + bf  (fp32 output)
// ---------------------------------------------------------------------------
__global__ void k_final(const __bf16* __restrict__ X, const __bf16* __restrict__ W,
                        const float* __restrict__ bias, float* __restrict__ out) {
  const int mt = blockIdx.x, nt = blockIdx.y, lane = threadIdx.x;
  const int m0 = mt * 16, n0 = nt * 16;

  v8f acc = {0.f, 0.f, 0.f, 0.f, 0.f, 0.f, 0.f, 0.f};
  for (int k = 0; k < D_MODEL; k += 32) {
    v16bf a = load_ab_tile(X + (size_t)m0 * D_MODEL + k, D_MODEL, lane);
    v16bf b = load_ab_tile(W + (size_t)n0 * D_MODEL + k, D_MODEL, lane);
    acc = wmma_bf16(a, b, acc);
  }
  const int n = n0 + (lane & 15);
  const float bv = bias[n];
  const int mbase = m0 + 8 * (lane >> 4);
#pragma unroll
  for (int j = 0; j < 8; ++j)
    out[(size_t)(mbase + j) * D_MODEL + n] = acc[j] + bv;
}

// ---------------------------------------------------------------------------
// Host launcher
// ---------------------------------------------------------------------------
extern "C" void kernel_launch(void* const* d_in, const int* in_sizes, int n_in,
                              void* d_out, int out_size, void* d_ws, size_t ws_size,
                              hipStream_t stream) {
  (void)in_sizes; (void)n_in; (void)out_size; (void)ws_size;

  const float* q  = (const float*)d_in[0];
  const float* k  = (const float*)d_in[1];
  const float* v  = (const float*)d_in[2];
  const float* Wq = (const float*)d_in[3];
  const float* bq = (const float*)d_in[4];
  const float* Wl = (const float*)d_in[5];
  const float* bl = (const float*)d_in[6];
  const float* Wp = (const float*)d_in[7];
  const float* bp = (const float*)d_in[8];
  const float* Wf = (const float*)d_in[9];
  const float* bf = (const float*)d_in[10];
  float* out = (float*)d_out;

  char* ws = (char*)d_ws;
  size_t off = 0;
  auto alloc = [&](size_t bytes) {
    void* p = ws + off;
    off = (off + bytes + 255) & ~(size_t)255;
    return p;
  };

  const size_t actE = (size_t)MROWS * D_MODEL;                 // 4096*768
  const size_t wE   = (size_t)D_MODEL * D_MODEL;               // 768*768
  const size_t hE   = (size_t)BATCH * NHEAD * SEQ * HDIM;      // == actE
  const size_t lE   = (size_t)BATCH * NHEAD * SEQ * SEQ;       // 50M
  const size_t sE   = (size_t)BATCH * NHEAD * SEQ;

  __bf16* qb   = (__bf16*)alloc(actE * 2);
  __bf16* kb   = (__bf16*)alloc(actE * 2);
  __bf16* vb   = (__bf16*)alloc(actE * 2);
  __bf16* Wqb  = (__bf16*)alloc(wE * 2);
  __bf16* Wfb  = (__bf16*)alloc(wE * 2);
  __bf16* qh   = (__bf16*)alloc(hE * 2);
  __bf16* kh   = (__bf16*)alloc(hE * 2);
  __bf16* vhT  = (__bf16*)alloc(hE * 2);
  __bf16* outh = (__bf16*)alloc(actE * 2);
  float*  Mst  = (float*)alloc(sE * 4);
  float*  Zst  = (float*)alloc(sE * 4);
  float*  Lmix = (float*)alloc(lE * 4);

  const int CV = 256;
  k_cvt_bf16<<<dim3((unsigned)((actE + CV - 1) / CV)), CV, 0, stream>>>(q,  qb,  (int)actE);
  k_cvt_bf16<<<dim3((unsigned)((actE + CV - 1) / CV)), CV, 0, stream>>>(k,  kb,  (int)actE);
  k_cvt_bf16<<<dim3((unsigned)((actE + CV - 1) / CV)), CV, 0, stream>>>(v,  vb,  (int)actE);
  k_cvt_bf16<<<dim3((unsigned)((wE   + CV - 1) / CV)), CV, 0, stream>>>(Wq, Wqb, (int)wE);
  k_cvt_bf16<<<dim3((unsigned)((wE   + CV - 1) / CV)), CV, 0, stream>>>(Wf, Wfb, (int)wE);

  dim3 gProj(MROWS / 16, D_MODEL / 16);   // (256, 48)
  k_proj<<<gProj, 32, 0, stream>>>(qb, Wqb, bq, qh,  0);
  k_proj<<<gProj, 32, 0, stream>>>(kb, Wqb, bq, kh,  1);
  k_proj<<<gProj, 32, 0, stream>>>(vb, Wqb, bq, vhT, 2);

  k_logits<<<dim3(SEQ / 16, SEQ / 16, BATCH), 32, 0, stream>>>(qh, kh, Wl, bl, Lmix);

  k_stats<<<dim3(SEQ / 16, NHEAD, BATCH), 256, 0, stream>>>(Lmix, Mst, Zst);

  k_attn<<<dim3(SEQ / 16, BATCH), 384, 0, stream>>>(Lmix, Mst, Zst, vhT, Wp, bp, outh);

  k_final<<<gProj, 32, 0, stream>>>(outh, Wfb, bf, out);
}
